// ShiftFuseZero_41437844472393
// MI455X (gfx1250) — matrix operand order repaired and checked by reading, not compiled, not run on hardware
//
#include <hip/hip_runtime.h>
#include <hip/hip_bf16.h>

typedef __attribute__((ext_vector_type(16))) _Float16 v16h;
typedef __attribute__((ext_vector_type(8)))  _Float16 v8h;
typedef __attribute__((ext_vector_type(8)))  float    v8f;

#define Bdim 64
#define Cdim 128
#define Tdim 256
#define Vdim 25
#define Kg   3            // number of graph kernels
#define TT   4            // t-values per block
#define MR   (TT * Vdim)  // 100 valid (tt,w) outputs
#define MP   (TT * 32)    // 128 padded M rows (w padded to 32 per tt)
#define KC   (Kg * Cdim)  // 384 = stage-2 reduction dim

// LDS layout (dynamic, no static LDS in main kernel):
//   [0      , 32768 )  xs : f16 [128][4][32]
//   [32768  , 131072)  wt : f16 [128][384]      (async-copied from global)
//   [131072 , 229376)  xk : f16 [128][384]      (aliased by yy f32 [128][128])
#define XS_OFF 0u
#define WT_OFF 32768u
#define XK_OFF 131072u
#define SMEM_BYTES 229376u

__device__ __forceinline__ float hswish(float z) {
    float t = fminf(fmaxf(z + 3.0f, 0.0f), 6.0f);
    return z * t * (1.0f / 6.0f);
}

// ---------------------------------------------------------------------------
// Prep: A_norm[k,v,w] = d[k,v]*|Af+Al|[k,v,w]*d[k,w],  d = rsqrt(max(rowsum,1e-6))
// AnT[k][w][v] (transposed, f16, padded 32x32; rows w>=25 are zero) and
// Wt[o][k*128+c] = W_gcn[k,o,c] / 3 (f16, [128][384]).  ~104 KB, L2-resident.
// ---------------------------------------------------------------------------
__global__ void __launch_bounds__(256)
prep_graph(const float* __restrict__ Af, const float* __restrict__ Al,
           const float* __restrict__ W, _Float16* __restrict__ AnT,
           _Float16* __restrict__ Wt) {
    __shared__ float dinv[Kg][Vdim];
    const int tid = threadIdx.x;
    if (tid < Kg * Vdim) {
        int k = tid / Vdim, v = tid % Vdim;
        float s = 0.f;
        for (int w = 0; w < Vdim; ++w)
            s += fabsf(Af[(k * Vdim + v) * Vdim + w] + Al[(k * Vdim + v) * Vdim + w]);
        dinv[k][v] = rsqrtf(fmaxf(s, 1e-6f));
    }
    __syncthreads();
    for (int idx = tid; idx < Kg * 32 * 32; idx += 256) {
        int k = idx / 1024, r = idx % 1024, w = r / 32, v = r % 32;
        float val = 0.f;
        if (w < Vdim && v < Vdim) {
            float a = fabsf(Af[(k * Vdim + v) * Vdim + w] + Al[(k * Vdim + v) * Vdim + w]);
            val = dinv[k][v] * a * dinv[k][w];
        }
        AnT[idx] = (_Float16)val;
    }
    for (int idx = tid; idx < Cdim * KC; idx += 256) {
        int o = idx / KC, kc = idx % KC, k = kc >> 7, c = kc & 127;
        Wt[idx] = (_Float16)(W[(k * Cdim + o) * Cdim + c] * (1.0f / 3.0f));
    }
}

// ---------------------------------------------------------------------------
// Main fused kernel: one block = (b, 4 consecutive t).  8 waves of wave32.
// ---------------------------------------------------------------------------
__global__ void __launch_bounds__(256)
gcn_fused(const float* __restrict__ x, const float* __restrict__ gamma,
          const float* __restrict__ beta, const _Float16* __restrict__ AnT,
          const _Float16* __restrict__ Wt, float* __restrict__ out) {
    extern __shared__ char smem[];
    _Float16* xs = (_Float16*)(smem + XS_OFF);  // [128][4][32]
    _Float16* wt = (_Float16*)(smem + WT_OFF);  // [128][384]
    _Float16* xk = (_Float16*)(smem + XK_OFF);  // [128][384]
    float*    yy = (float*)(smem + XK_OFF);     // [128][128], aliases xk

    const int tid  = threadIdx.x;
    const int lane = tid & 31;
    const int wave = tid >> 5;
    const int lrow = lane & 15;
    const int kbA  = (lane < 16) ? 0 : 8;   // A-fragment K sub-base (ISA layout)
    const int kbB  = (lane < 16) ? 0 : 16;  // unused; B fragments also use kbA
    const int blk  = blockIdx.x;
    const int b    = blk >> 6;
    const int t0   = (blk & 63) * TT;

    // ---- async-copy Wt (96 KB) global -> LDS; tracked by ASYNCcnt ----------
    {
        const unsigned ldsbase = __builtin_amdgcn_groupstaticsize() + WT_OFF;
        const char* wsrc = (const char*)Wt;
        for (int ch = tid; ch < Cdim * KC * 2 / 16; ch += 256) {  // 24 iters
            unsigned loff = ldsbase + (unsigned)ch * 16u;
            unsigned long long ga = (unsigned long long)(uintptr_t)(wsrc + ch * 16);
            asm volatile("global_load_async_to_lds_b128 %0, %1, off"
                         :: "v"(loff), "v"(ga) : "memory");
        }
    }

    // ---- load x slab -> xs (f16), zero-pad v=25..31 ------------------------
    for (int s = tid; s < Cdim * TT * 32; s += 256) {
        int v = s & 31, tt = (s >> 5) & 3, c = s >> 7;
        float val = 0.f;
        if (v < Vdim)
            val = x[((b * Cdim + c) * Tdim + (t0 + tt)) * Vdim + v];
        xs[s] = (_Float16)val;
    }

    asm volatile("s_wait_asynccnt 0" ::: "memory");  // Wt in LDS (per wave)
    __syncthreads();                                  // ... for all waves

    // ---- stage 1: xk[tt*32+w][k*128+c] = sum_v AnT[k][w][v]*xs[c][tt][v] ---
    // Rows w=25..31 of AnT are zero, so all 32 w-rows store unconditionally.
    for (int j = wave; j < Kg * TT * 8; j += 8) {
        int k = j >> 5, r = j & 31, tt = r >> 3, nt = r & 7;
        int c = nt * 16 + lrow;
        v8h blo = *(const v8h*)&xs[(c * TT + tt) * 32 + kbA];
        v8h bhi = *(const v8h*)&xs[(c * TT + tt) * 32 + kbA + 16];
        v16h Bf = __builtin_shufflevector(blo, bhi, 0,1,2,3,4,5,6,7,8,9,10,11,12,13,14,15);
        v8f d0 = {}, d1 = {};
        {
            const _Float16* arow = AnT + (k * 32 + lrow) * 32;       // w = lrow
            v8h lo = *(const v8h*)(arow + kbA);
            v8h hi = *(const v8h*)(arow + kbA + 16);
            v16h Afr = __builtin_shufflevector(lo, hi, 0,1,2,3,4,5,6,7,8,9,10,11,12,13,14,15);
            d0 = __builtin_amdgcn_wmma_f32_16x16x32_f16(false, Afr, false, Bf,
                                                        (short)0, d0, false, false);
        }
        {
            const _Float16* arow = AnT + (k * 32 + 16 + lrow) * 32;  // w = 16+lrow
            v8h lo = *(const v8h*)(arow + kbA);
            v8h hi = *(const v8h*)(arow + kbA + 16);
            v16h Afr = __builtin_shufflevector(lo, hi, 0,1,2,3,4,5,6,7,8,9,10,11,12,13,14,15);
            d1 = __builtin_amdgcn_wmma_f32_16x16x32_f16(false, Afr, false, Bf,
                                                        (short)0, d1, false, false);
        }
        int col = k * Cdim + nt * 16 + lrow;
#pragma unroll
        for (int i = 0; i < 8; ++i) {
            int m = (lane < 16) ? i : i + 8;          // D row per ISA C/D layout
            xk[(tt * 32 + m) * KC + col]      = (_Float16)d0[i];  // w = m
            xk[(tt * 32 + 16 + m) * KC + col] = (_Float16)d1[i];  // w = 16+m
        }
    }
    __syncthreads();

    // ---- stage 2: D[o][m] = sum_kc Wt[o][kc] * xk[m][kc] -------------------
    // A = Wt (M=o), B = xk^T (N=m).  Each wave owns one o-tile (Mt = wave),
    // so A-fragments are loaded once per K-chunk and reused across 8 N-tiles.
    v8f acc[8] = {};
#pragma unroll
    for (int q = 0; q < KC / 32; ++q) {
        const _Float16* arow = wt + (wave * 16 + lrow) * KC + q * 32;
        v8h lo = *(const v8h*)(arow + kbA);
        v8h hi = *(const v8h*)(arow + kbA + 16);
        v16h Afr = __builtin_shufflevector(lo, hi, 0,1,2,3,4,5,6,7,8,9,10,11,12,13,14,15);
#pragma unroll
        for (int i = 0; i < 8; ++i) {
            v8h blo = *(const v8h*)&xk[(i * 16 + lrow) * KC + q * 32 + kbA];
            v8h bhi = *(const v8h*)&xk[(i * 16 + lrow) * KC + q * 32 + kbA + 16];
            v16h Bf = __builtin_shufflevector(blo, bhi, 0,1,2,3,4,5,6,7,8,9,10,11,12,13,14,15);
            acc[i] = __builtin_amdgcn_wmma_f32_16x16x32_f16(false, Afr, false, Bf,
                                                            (short)0, acc[i], false, false);
        }
    }
    __syncthreads();  // all xk reads done; yy may now overwrite that LDS

    // park accumulators as yy[o][m]; lanes write consecutive m (conflict-free)
#pragma unroll
    for (int i = 0; i < 8; ++i) {
        int m = i * 16 + lrow;
#pragma unroll
        for (int r = 0; r < 8; ++r) {
            int o = wave * 16 + ((lane < 16) ? r : r + 8);
            yy[o * MP + m] = acc[i][r];
        }
    }
    __syncthreads();

    // ---- epilogue: BN -> hardswish -> +x -> hardswish ----------------------
    const float bnr = rsqrtf(1.0f + 1e-5f);
    for (int s = tid; s < Cdim * MR; s += 256) {
        int m100 = s % MR, o = s / MR;
        int tt = m100 / Vdim, w = m100 % Vdim;
        float y = yy[o * MP + tt * 32 + w];
        y = y * (gamma[o] * bnr) + beta[o];
        y = hswish(y);
        int gi = ((b * Cdim + o) * Tdim + (t0 + tt)) * Vdim + w;
        out[gi] = hswish(y + x[gi]);   // residual re-read hits L0/L2 (just loaded)
    }
}

extern "C" void kernel_launch(void* const* d_in, const int* in_sizes, int n_in,
                              void* d_out, int out_size, void* d_ws, size_t ws_size,
                              hipStream_t stream) {
    const float* x  = (const float*)d_in[0];
    const float* Af = (const float*)d_in[1];
    const float* Al = (const float*)d_in[2];
    const float* W  = (const float*)d_in[3];
    const float* g  = (const float*)d_in[4];
    const float* be = (const float*)d_in[5];

    _Float16* AnT = (_Float16*)d_ws;                       // 3*32*32 f16 = 6 KB
    _Float16* Wt  = (_Float16*)((char*)d_ws + 8192);       // 128*384 f16 = 96 KB

    prep_graph<<<1, 256, 0, stream>>>(Af, Al, W, AnT, Wt);

    gcn_fused<<<Bdim * (Tdim / TT), 256, SMEM_BYTES, stream>>>(x, g, be, AnT, Wt,
                                                               (float*)d_out);
}
